// MultiHeadAttention_3075196584284
// MI455X (gfx1250) — compile-verified, matrix operand order checked by reading, not run on hardware
//
#include <hip/hip_runtime.h>
#include <hip/hip_bf16.h>

// ---------------------------------------------------------------------------
// MultiHeadAttention with 1.5-entmax, CDNA5 (gfx1250), wave32 WMMA bf16 path.
// B=4, S=1024, INPUTS=1024, HEADS=16, HID=64.
//
// Pipeline (all on `stream`, deterministic):
//   cvt(x)->bf16, cvt(W)->bf16, proj_gemm  (x3 for Q/K/V; V stored transposed)
//   fused scores -> entmax1.5 -> attn@V
//   cvt(attn_out)->bf16, cvt(Wo)->bf16, proj_gemm -> f32 output
//
// proj_gemm stages its 16x1024 A panel into LDS via the Tensor Data Mover
// (tensor_load_to_lds + s_wait_tensorcnt) and shares it across all 8 waves.
// ---------------------------------------------------------------------------

typedef __attribute__((ext_vector_type(16))) __bf16          v16bf;
typedef __attribute__((ext_vector_type(8)))  float           v8f;
typedef __attribute__((ext_vector_type(16))) unsigned short  v16u;
typedef __attribute__((ext_vector_type(8)))  unsigned short  u16x8;
typedef __attribute__((ext_vector_type(4)))  unsigned int    u32x4;
typedef __attribute__((ext_vector_type(8)))  int             i32x8;
typedef __attribute__((ext_vector_type(4)))  int             i32x4;

#define WMMA_BF16(a, b, c) \
    __builtin_amdgcn_wmma_f32_16x16x32_bf16(false, (a), false, (b), (short)0, (c), false, false)

__device__ __forceinline__ unsigned short f2b(float f) {
    __hip_bfloat16 h = __float2bfloat16(f);
    return __builtin_bit_cast(unsigned short, h);
}

// 16 bf16 fragment elements for one lane from bf16 memory (global or LDS).
// CDNA5 16-bit 16x32 A layout: elems 0..7 = k..k+7, elems 8..15 = k+16..k+23.
__device__ __forceinline__ v16bf ld16b(const unsigned short* __restrict__ p) {
    u16x8 a = *(const u16x8*)p;          // k .. k+7
    u16x8 b = *(const u16x8*)(p + 16);   // k+16 .. k+23
    v16u u;
#pragma unroll
    for (int i = 0; i < 8; ++i) { u[i] = a[i]; u[8 + i] = b[i]; }
    return __builtin_bit_cast(v16bf, u);
}

// ---------------------------------------------------------------------------
// TDM: DMA a 2-D bf16 tile (rows x cols, row stride in elements) into LDS.
// D# layout per CDNA5 ISA ch.8 (groups 0/1; groups 2/3 zero -> 2-D tensor).
// Wave-level op (EXEC ignored); caller waits TENSORcnt and barriers.
// ---------------------------------------------------------------------------
__device__ __forceinline__ void tdm_load_tile_2d(unsigned lds_off, const void* gptr,
                                                 unsigned cols, unsigned rows,
                                                 unsigned row_stride_elems) {
    const unsigned long long ga = (unsigned long long)gptr;
    u32x4 g0;
    g0[0] = 1u;                                           // count=1, user descriptor
    g0[1] = lds_off;                                      // lds_addr (bytes)
    g0[2] = (unsigned)(ga & 0xFFFFFFFFu);                 // global_addr[31:0]
    g0[3] = (unsigned)((ga >> 32) & 0x01FFFFFFu)          // global_addr[56:32]
            | (2u << 30);                                 // type = 2 ("image")
    i32x8 g1;
    g1[0] = (int)(1u << 16);                              // mask=0, data_size=1 (2 B)
    g1[1] = (int)((cols & 0xFFFFu) << 16);                // tensor_dim0[15:0]
    g1[2] = (int)(((cols >> 16) & 0xFFFFu)                // tensor_dim0[31:16]
            | ((rows & 0xFFFFu) << 16));                  // tensor_dim1[15:0]
    g1[3] = (int)(((rows >> 16) & 0xFFFFu)                // tensor_dim1[31:16]
            | ((cols & 0xFFFFu) << 16));                  // tile_dim0
    g1[4] = (int)(rows & 0xFFFFu);                        // tile_dim1 (tile_dim2 = 0)
    g1[5] = (int)row_stride_elems;                        // tensor_dim0_stride[31:0]
    g1[6] = 0;                                            // stride hi / dim1_stride lo
    g1[7] = 0;                                            // dim1_stride hi
    const i32x4 z4 = {0, 0, 0, 0};
#if __clang_major__ >= 23
    const i32x8 z8 = {0, 0, 0, 0, 0, 0, 0, 0};
    __builtin_amdgcn_tensor_load_to_lds(g0, g1, z4, z4, z8, 0);
#else
    __builtin_amdgcn_tensor_load_to_lds(g0, g1, z4, z4, 0);
#endif
}

// ---------------------------------------------------------------------------
// Streaming f32 -> bf16 conversion, 8 elements/thread, b128 in / b128 out.
// ---------------------------------------------------------------------------
__global__ void cvt_f32_bf16(const float* __restrict__ in, unsigned short* __restrict__ out,
                             int n8) {
    const int i = blockIdx.x * 256 + threadIdx.x;
    if (i >= n8) return;
    const float4* p = (const float4*)(in + (size_t)i * 8);
    float4 a = p[0], b = p[1];
    u16x8 o;
    o[0] = f2b(a.x); o[1] = f2b(a.y); o[2] = f2b(a.z); o[3] = f2b(a.w);
    o[4] = f2b(b.x); o[5] = f2b(b.y); o[6] = f2b(b.z); o[7] = f2b(b.w);
    *(u16x8*)(out + (size_t)i * 8) = o;
}

// ---------------------------------------------------------------------------
// Projection GEMM on pre-converted bf16: Y = X @ W^T + bias.
// X:[4096,1024] bf16, W:[1024,1024] bf16 (row-major [out,in] -> B column = W row).
// A panel (16 rows x 1024 k, 32 KB) is TDM-DMA'd to LDS once per block and
// shared by all 8 waves; each wave owns a 16x64 strip (4 accumulators).
// Grid: (M/16, N/512), block 256.
// MODE 0: store bf16 [B,H,S,64]   (Q, K)
// MODE 1: store bf16 [B,H,64,S]   (V transposed: attn@V B-operand contiguous in k)
// MODE 2: store f32  [4096,1024]  (final output)
// ---------------------------------------------------------------------------
template <int MODE>
__global__ void proj_gemm(const unsigned short* __restrict__ X,
                          const unsigned short* __restrict__ W,
                          const float* __restrict__ bias, void* __restrict__ out) {
    __shared__ __align__(16) unsigned short sA[16 * 1024];   // 32 KB A panel

    const int l    = threadIdx.x & 31;
    const int wave = threadIdx.x >> 5;
    const int l15  = l & 15;
    const int hi   = l >> 4;

    const int m0 = blockIdx.x * 16;
    const int nb = blockIdx.y * 512 + wave * 64;          // wave's 64-col strip

    if (wave == 0) {
        const unsigned lds_off = (unsigned)(unsigned long long)(const void*)sA;
        tdm_load_tile_2d(lds_off, X + (size_t)m0 * 1024, 1024, 16, 1024);
        __builtin_amdgcn_s_wait_tensorcnt(0);
    }
    __syncthreads();

    const unsigned short* ap  = sA + (size_t)l15 * 1024;  // A row in LDS
    const unsigned short* bp0 = W + (size_t)(nb + 0 * 16 + l15) * 1024;
    const unsigned short* bp1 = W + (size_t)(nb + 1 * 16 + l15) * 1024;
    const unsigned short* bp2 = W + (size_t)(nb + 2 * 16 + l15) * 1024;
    const unsigned short* bp3 = W + (size_t)(nb + 3 * 16 + l15) * 1024;

    v8f acc0 = {}, acc1 = {}, acc2 = {}, acc3 = {};
#pragma unroll 2
    for (int kb = 0; kb < 1024; kb += 32) {
        const int ks = kb + hi * 8;
        if (kb + 64 < 1024) __builtin_prefetch(bp0 + ks + 64, 0, 0);  // global_prefetch_b8
        const v16bf a = ld16b(ap + ks);                    // ds_read_b128 x2
        acc0 = WMMA_BF16(a, ld16b(bp0 + ks), acc0);
        acc1 = WMMA_BF16(a, ld16b(bp1 + ks), acc1);
        acc2 = WMMA_BF16(a, ld16b(bp2 + ks), acc2);
        acc3 = WMMA_BF16(a, ld16b(bp3 + ks), acc3);
    }

#pragma unroll
    for (int t = 0; t < 4; ++t) {
        const v8f acc = (t == 0) ? acc0 : (t == 1) ? acc1 : (t == 2) ? acc2 : acc3;
        const int n = nb + t * 16 + l15;
        const float bv = bias[n];
#pragma unroll
        for (int r = 0; r < 8; ++r) {
            const int mr = m0 + r + 8 * hi;               // D row for this element
            const float v = acc[r] + bv;
            if (MODE == 2) {
                ((float*)out)[(size_t)mr * 1024 + n] = v;
            } else {
                const int bb = mr >> 10, s = mr & 1023;   // (batch, seq)
                const int h = n >> 6, d = n & 63;         // (head, hid)
                unsigned short* o = (unsigned short*)out;
                if (MODE == 0)
                    o[((size_t)(bb * 16 + h) * 1024 + s) * 64 + d] = f2b(v);
                else
                    o[((size_t)(bb * 16 + h) * 64 + d) * 1024 + s] = f2b(v);
            }
        }
    }
}

// ---------------------------------------------------------------------------
// Fused scores -> entmax1.5 -> attn@V.  One workgroup per (b, h, 16 query rows).
// ---------------------------------------------------------------------------
__global__ void attn_entmax_kernel(const unsigned short* __restrict__ Qh,
                                   const unsigned short* __restrict__ Kh,
                                   const unsigned short* __restrict__ Vt,
                                   const int* __restrict__ mask,
                                   float* __restrict__ attn_out) {
    __shared__ __align__(16) float sc[16 * 1024];   // x = masked scores * (1/8) * (1/2)
    __shared__ __align__(16) float buf[8 * 1024];   // bitonic scratch; reused as bf16 attn
    __shared__ float rmax[16];
    __shared__ float tstar[16];
    __shared__ float red[16][17];

    const int bid = blockIdx.x;
    const int qt = bid & 63, h = (bid >> 6) & 15, b = bid >> 10;
    const int tid = threadIdx.x;
    const int l = tid & 31, wave = tid >> 5;
    const int l15 = l & 15, hi = l >> 4;

    const unsigned short* Qbase = Qh + ((size_t)(b * 16 + h) * 1024 + qt * 16) * 64;
    const unsigned short* Kbase = Kh + (size_t)(b * 16 + h) * 1024 * 64;
    const unsigned short* Vbase = Vt + (size_t)(b * 16 + h) * 64 * 1024;

    // ---- Phase 1: scores = Q K^T / sqrt(64), masked, pre-scaled by 1/2 ----
    const v16bf qa0 = ld16b(Qbase + l15 * 64 + hi * 8);        // k = 0..31 chunk
    const v16bf qa1 = ld16b(Qbase + l15 * 64 + 32 + hi * 8);   // k = 32..63 chunk

#pragma unroll 2
    for (int i = 0; i < 8; ++i) {
        const int nt = wave * 8 + i;               // 16-key column tile
        const int key = nt * 16 + l15;             // this lane's key column
        const unsigned short* kp = Kbase + (size_t)key * 64;
        v8f acc = {};
        acc = WMMA_BF16(qa0, ld16b(kp + hi * 8), acc);
        acc = WMMA_BF16(qa1, ld16b(kp + 32 + hi * 8), acc);
        const bool live = (mask[b * 1024 + key] != 0);
#pragma unroll
        for (int r = 0; r < 8; ++r) {
            const float v = live ? acc[r] * 0.0625f : -5.0e11f;  // (1/8)*(1/2), -1e12/2
            sc[(r + 8 * hi) * 1024 + key] = v;
        }
    }
    __syncthreads();

    // ---- Phase 2: per-row max ----
    {
        const int row = tid >> 4, seg = tid & 15;
        float m = -3.4e38f;
        for (int c = seg * 64; c < seg * 64 + 64; ++c) m = fmaxf(m, sc[row * 1024 + c]);
        red[row][seg] = m;
    }
    __syncthreads();
    if (tid < 16) {
        float m = -3.4e38f;
        for (int i = 0; i < 16; ++i) m = fmaxf(m, red[tid][i]);
        rmax[tid] = m;
    }
    __syncthreads();

    // ---- Phase 3: exact entmax1.5 (sort-based), 8 rows per batch ----
    for (int rb = 0; rb < 2; ++rb) {
        for (int t = tid; t < 8 * 1024; t += 256) {
            const int rl = t >> 10, c = t & 1023;
            buf[t] = sc[(rb * 8 + rl) * 1024 + c] - rmax[rb * 8 + rl];
        }
        __syncthreads();

        // Bitonic sort, descending, 8 rows of 1024 in parallel.
        for (unsigned k2 = 2; k2 <= 1024; k2 <<= 1) {
            for (unsigned j = k2 >> 1; j > 0; j >>= 1) {
                for (unsigned t = tid; t < 8 * 1024; t += 256) {
                    const unsigned rl = t >> 10, ii = t & 1023;
                    const unsigned jj = ii ^ j;
                    if (jj > ii) {
                        float* z = buf + rl * 1024;
                        const float x = z[ii], y = z[jj];
                        const bool desc = ((ii & k2) == 0);
                        if (desc ? (x < y) : (x > y)) { z[ii] = y; z[jj] = x; }
                    }
                }
                __syncthreads();
            }
        }

        // Serial tau scan per row (predicate tau<=z_k is a true-prefix -> last hit).
        if (tid < 8) {
            const float* z = buf + tid * 1024;
            float sum = 0.f, sumsq = 0.f, ts = 0.f;
            for (int kk = 1; kk <= 1024; ++kk) {
                const float zk = z[kk - 1];
                sum += zk; sumsq += zk * zk;
                const float fk = (float)kk;
                const float mean = sum / fk;
                const float ss = sumsq - sum * mean;       // k*(cumsq/k - mean^2)
                const float delta = (1.0f - ss) / fk;
                const float tau = mean - sqrtf(fmaxf(delta, 0.f));
                if (tau <= zk) ts = tau;
            }
            tstar[rb * 8 + tid] = ts;
        }
        __syncthreads();
    }

    // ---- Phase 4: attn = clip(x - rmax - tau)^2 as bf16 into reused buffer ----
    unsigned short* attnb = (unsigned short*)buf;          // 16x1024 bf16 == 32KB
    for (int t = tid; t < 16 * 1024; t += 256) {
        const int row = t >> 10, c = t & 1023;
        float p = sc[row * 1024 + c] - rmax[row] - tstar[row];
        p = fmaxf(p, 0.f);
        attnb[row * 1024 + c] = f2b(p * p);
    }
    __syncthreads();

    // ---- Phase 5: out_tile[16 x 64] = attn @ V  (waves 0..3, one 16-col tile each)
    if (wave < 4) {
        const int d = wave * 16 + l15;
        const unsigned short* vp = Vbase + (size_t)d * 1024;     // contiguous along key
        const unsigned short* arow = attnb + l15 * 1024;
        v8f acc = {};
#pragma unroll 4
        for (int kb = 0; kb < 1024; kb += 32) {
            v16bf a = ld16b(arow + kb + hi * 8);
            v16bf bm = ld16b(vp + kb + hi * 8);
            acc = WMMA_BF16(a, bm, acc);
        }
#pragma unroll
        for (int r = 0; r < 8; ++r) {
            const int s = qt * 16 + r + 8 * hi;
            attn_out[((size_t)b * 1024 + s) * 1024 + h * 64 + d] = acc[r];  // [B,S,H*64]
        }
    }
}

// ---------------------------------------------------------------------------
extern "C" void kernel_launch(void* const* d_in, const int* in_sizes, int n_in,
                              void* d_out, int out_size, void* d_ws, size_t ws_size,
                              hipStream_t stream) {
    (void)in_sizes; (void)n_in; (void)out_size; (void)ws_size;
    const float* q    = (const float*)d_in[0];
    const float* k    = (const float*)d_in[1];
    const float* v    = (const float*)d_in[2];
    const int*   mask = (const int*)  d_in[3];
    const float* Wq = (const float*)d_in[4];  const float* bq = (const float*)d_in[5];
    const float* Wk = (const float*)d_in[6];  const float* bk = (const float*)d_in[7];
    const float* Wv = (const float*)d_in[8];  const float* bv = (const float*)d_in[9];
    const float* Wo = (const float*)d_in[10]; const float* bo = (const float*)d_in[11];

    char* ws = (char*)d_ws;
    unsigned short* Qh = (unsigned short*)(ws);                    //  8 MiB bf16 [B,H,S,64]
    unsigned short* Kh = (unsigned short*)(ws + (8u  << 20));      //  8 MiB bf16 [B,H,S,64]
    unsigned short* Vt = (unsigned short*)(ws + (16u << 20));      //  8 MiB bf16 [B,H,64,S]
    float*          AO = (float*)         (ws + (24u << 20));      // 16 MiB f32  [B,S,1024]
    unsigned short* Xb = (unsigned short*)(ws + (40u << 20));      //  8 MiB bf16 X staging
    unsigned short* Wb = (unsigned short*)(ws + (48u << 20));      //  2 MiB bf16 W staging

    const int nX8 = 4096 * 1024 / 8;   // X elements / 8
    const int nW8 = 1024 * 1024 / 8;   // W elements / 8
    dim3 cg_x(nX8 / 256), cg_w(nW8 / 256);
    dim3 gg(256, 2), blk(256);

    cvt_f32_bf16<<<cg_x, blk, 0, stream>>>(q, Xb, nX8);
    cvt_f32_bf16<<<cg_w, blk, 0, stream>>>(Wq, Wb, nW8);
    proj_gemm<0><<<gg, blk, 0, stream>>>(Xb, Wb, bq, Qh);

    cvt_f32_bf16<<<cg_x, blk, 0, stream>>>(k, Xb, nX8);
    cvt_f32_bf16<<<cg_w, blk, 0, stream>>>(Wk, Wb, nW8);
    proj_gemm<0><<<gg, blk, 0, stream>>>(Xb, Wb, bk, Kh);

    cvt_f32_bf16<<<cg_x, blk, 0, stream>>>(v, Xb, nX8);
    cvt_f32_bf16<<<cg_w, blk, 0, stream>>>(Wv, Wb, nW8);
    proj_gemm<1><<<gg, blk, 0, stream>>>(Xb, Wb, bv, Vt);

    attn_entmax_kernel<<<dim3(4096), blk, 0, stream>>>(Qh, Kh, Vt, mask, AO);

    cvt_f32_bf16<<<cg_x, blk, 0, stream>>>(AO, Xb, nX8);
    cvt_f32_bf16<<<cg_w, blk, 0, stream>>>(Wo, Wb, nW8);
    proj_gemm<2><<<gg, blk, 0, stream>>>(Xb, Wb, bo, (float*)d_out);
}
